// ParallelT5Attention_17042430231366
// MI455X (gfx1250) — matrix operand order repaired and checked by reading, not compiled, not run on hardware
//
#include <hip/hip_runtime.h>
#include <stdint.h>

// ---------------------------------------------------------------------------
// T5 attention (B=4, S=1024, D=1024, H=16, dkv=64) for gfx1250 / MI455X.
// All matmuls via v_wmma_f32_16x16x32_bf16 (bf16 operands, fp32 accumulate).
// GEMMs: LDS-staged weight tiles (double-buffered) shared by all 4 waves of
// a block; A fragments register double-buffered. Flash attention with K/V
// fragment double-buffering.
// ---------------------------------------------------------------------------

#define SEQ     1024
#define DMODEL  1024
#define NHEADS  16
#define DKV     64
#define NBATCH  4

// LDS weight tile: 64 rows (n) x 32 k-elements, padded row stride 40 elements
// (80 bytes) -> fragment reads hit 16 distinct banks across lanes.
#define BROW    40

typedef __attribute__((ext_vector_type(16))) __bf16 v16bf;
typedef __attribute__((ext_vector_type(8)))  float  v8f;

union AFrag {
  uint4          q[2];   // two 16-byte chunks per lane
  v16bf          v;
  unsigned short us[16];
};

__device__ __forceinline__ unsigned short f2bf(float x) {
  union { float f; unsigned u; } t; t.f = x;
  unsigned u = t.u;
  unsigned r = (u + 0x7FFFu + ((u >> 16) & 1u)) >> 16;   // RNE
  return (unsigned short)r;
}

__device__ __forceinline__ v8f wmma_bf16(const AFrag& a, const AFrag& b, v8f c) {
  // (neg_a, A, neg_b, B, c_mod, C, reuse_a, reuse_b)
  return __builtin_amdgcn_wmma_f32_16x16x32_bf16(false, a.v, false, b.v,
                                                 (short)0, c, false, false);
}

// --------------------------- conversion kernels ----------------------------

__global__ __launch_bounds__(256) void cvt_bf16_kernel(
    const float* __restrict__ in, unsigned short* __restrict__ out, int n) {
  int i = blockIdx.x * 256 + threadIdx.x;
  if (i < n) out[i] = f2bf(in[i]);
}

// W [1024,1024] fp32 -> Wt [n][k] bf16 (Wt[n*1024+k] = W[k*1024+n]).
// LDS-tiled 32x32 so both sides are coalesced. Block (32,8), grid (32,32).
__global__ __launch_bounds__(256) void transpose_bf16_kernel(
    const float* __restrict__ in, unsigned short* __restrict__ out) {
  __shared__ unsigned short tile[32][33];
  const int tx = threadIdx.x;          // 0..31
  const int ty = threadIdx.y;          // 0..7
  const int bk = blockIdx.y * 32;      // k-range of this tile
  const int bn = blockIdx.x * 32;      // n-range of this tile
#pragma unroll
  for (int i = 0; i < 4; ++i) {
    int k = bk + ty + 8 * i;
    tile[ty + 8 * i][tx] = f2bf(in[(size_t)k * DMODEL + bn + tx]);
  }
  __syncthreads();
#pragma unroll
  for (int i = 0; i < 4; ++i) {
    int n = bn + ty + 8 * i;
    out[(size_t)n * DMODEL + bk + tx] = tile[tx][ty + 8 * i];
  }
}

// T5 relative position bias: bias_d[h][d + 1023], d = k - q in [-1023,1023]
__global__ __launch_bounds__(256) void bias_kernel(
    const float* __restrict__ rel_bias, float* __restrict__ bias_d) {
  int i = blockIdx.x * 256 + threadIdx.x;
  if (i >= NHEADS * 2047) return;
  int h = i / 2047;
  int idx = i % 2047;
  int d = idx - 1023;                      // key_pos - query_pos
  int bucket = (d > 0) ? 16 : 0;           // bidirectional: num_buckets/2 = 16
  int rel = d < 0 ? -d : d;
  int bval;
  if (rel < 8) {
    bval = rel;                            // max_exact = 8
  } else {
    float l = logf((float)rel * (1.0f / 8.0f)) * (1.0f / logf(16.0f)) * 8.0f;
    bval = 8 + (int)l;
    if (bval > 15) bval = 15;
  }
  bucket += bval;
  bias_d[h * 2047 + idx] = rel_bias[bucket * NHEADS + h];  // table [32][16]
}

// ------------------------------- QKV GEMM ----------------------------------
// C[4096,1024] = Xbf @ W (W given transposed bf16). Block = 4 waves sharing
// one 64-col weight strip; wave computes a 32x64 tile. Weight K-tiles are
// cooperatively staged into LDS (double-buffered); A fragments are register
// double-buffered from global.
// sel 0 -> Q [B,H,S,dkv], sel 1 -> K [B,H,S,dkv], sel 2 -> Vt [B,H,dkv,S].

__global__ __launch_bounds__(128) void gemm_qkv_kernel(
    const unsigned short* __restrict__ Xbf,
    const unsigned short* __restrict__ Wqt,
    const unsigned short* __restrict__ Wkt,
    const unsigned short* __restrict__ Wvt,
    unsigned short* __restrict__ Qout,
    unsigned short* __restrict__ Kout,
    unsigned short* __restrict__ VtOut) {
  __shared__ __align__(16) unsigned short Blds[2][64 * BROW];

  const int tid = threadIdx.x;
  const int wave = tid >> 5;
  const int lane = tid & 31;
  const int hh = lane >> 4, ln = lane & 15;
  const int m0 = (blockIdx.x * 4 + wave) * 32;
  const int head = blockIdx.y;
  const int n0 = head * 64;
  const int sel = blockIdx.z;
  const unsigned short* Wt = (sel == 0) ? Wqt : ((sel == 1) ? Wkt : Wvt);

  // cooperative stage mapping: 2 threads per row, 2 x uint4 (16 elems) each
  const int srow = tid >> 1;                 // 0..63
  const int sj = (tid & 1) * 16;             // element offset within k-tile
  const unsigned short* wsrc = Wt + (size_t)(n0 + srow) * DMODEL + sj;
  unsigned short* wdst0 = &Blds[0][srow * BROW + sj];
  unsigned short* wdst1 = &Blds[1][srow * BROW + sj];

  const v8f vzero = {0.f, 0.f, 0.f, 0.f, 0.f, 0.f, 0.f, 0.f};
  v8f acc[2][4];
#pragma unroll
  for (int mi = 0; mi < 2; ++mi)
#pragma unroll
    for (int t = 0; t < 4; ++t) acc[mi][t] = vzero;

  const unsigned short* xr[2] = {
      Xbf + (size_t)(m0 + ln) * DMODEL,
      Xbf + (size_t)(m0 + 16 + ln) * DMODEL};

  AFrag a[2][2];   // [buf][mtile], register double-buffered
  uint4 st0, st1;  // weight staging registers

  auto loadA = [&](int buf, int k0) {
#pragma unroll
    for (int mi = 0; mi < 2; ++mi) {
      a[buf][mi].q[0] = *(const uint4*)(xr[mi] + k0 + 16 * hh);
      a[buf][mi].q[1] = *(const uint4*)(xr[mi] + k0 + 16 * hh + 8);
    }
  };
  auto gloadW = [&](int k0) {
    st0 = *(const uint4*)(wsrc + k0);
    st1 = *(const uint4*)(wsrc + k0 + 8);
  };
  auto storeW = [&](unsigned short* dst) {
    *(uint4*)(dst) = st0;
    *(uint4*)(dst + 8) = st1;
  };

  // prologue: stage k-tile 0
  gloadW(0);
  loadA(0, 0);
  storeW(wdst0);
  __syncthreads();

  for (int k0 = 0; k0 < DMODEL; k0 += 32) {
    const int cur = (k0 >> 5) & 1;
    const int nxt = cur ^ 1;
    const bool more = (k0 + 32) < DMODEL;
    if (more) {                 // issue next-tile global loads early
      gloadW(k0 + 32);
      loadA(nxt, k0 + 32);
    }
    // compute from LDS buffer `cur`
#pragma unroll
    for (int t = 0; t < 4; ++t) {
      AFrag b;
      const unsigned short* br = &Blds[cur][(t * 16 + ln) * BROW + 16 * hh];
      b.q[0] = *(const uint4*)(br);
      b.q[1] = *(const uint4*)(br + 8);
#pragma unroll
      for (int mi = 0; mi < 2; ++mi)
        acc[mi][t] = wmma_bf16(a[cur][mi], b, acc[mi][t]);
    }
    __syncthreads();            // everyone done reading `cur`
    if (more) {
      storeW(nxt ? wdst1 : wdst0);
      __syncthreads();          // buffer `nxt` visible to all waves
    }
  }

  const int bb = m0 >> 10;   // batch index (32 rows never cross a batch)
#pragma unroll
  for (int mi = 0; mi < 2; ++mi) {
#pragma unroll
    for (int t = 0; t < 4; ++t) {
#pragma unroll
      for (int r = 0; r < 8; ++r) {
        const int row = m0 + 16 * mi + r + 8 * hh;
        const int s = row & (SEQ - 1);
        const int d = t * 16 + ln;
        const unsigned short val = f2bf(acc[mi][t][r]);
        if (sel == 2) {
          VtOut[((size_t)((bb * NHEADS + head) * DKV + d)) * SEQ + s] = val;
        } else {
          unsigned short* O = (sel == 0) ? Qout : Kout;
          O[((size_t)((bb * NHEADS + head) * SEQ + s)) * DKV + d] = val;
        }
      }
    }
  }
}

// ----------------------------- attention -----------------------------------
// One wave per (b, h, 16 query rows). Flash-style online softmax over k-tiles
// of 32, with K/V fragments double-buffered so next-tile loads overlap the
// softmax VALU work and the LDS layout round-trip.
// C-fragment element r on lane L is (M = r + 8*(L>>4), N = L&15).

__global__ __launch_bounds__(32) void attn_kernel(
    const unsigned short* __restrict__ Q,
    const unsigned short* __restrict__ K,
    const unsigned short* __restrict__ Vt,
    const float* __restrict__ bias_d,
    const float* __restrict__ mask,
    unsigned short* __restrict__ AttnOut) {
  __shared__ __align__(16) unsigned short Plds[16 * 32];

  const int lane = threadIdx.x;
  const int hh = lane >> 4;
  const int ln = lane & 15;

  const int qtile = blockIdx.x & 63;
  const int bh = blockIdx.x >> 6;
  const int b = bh >> 4;
  const int h = bh & 15;
  const int q0 = qtile * 16;

  const unsigned short* Qb = Q  + (size_t)(b * NHEADS + h) * SEQ * DKV;
  const unsigned short* Kb = K  + (size_t)(b * NHEADS + h) * SEQ * DKV;
  const unsigned short* Vb = Vt + (size_t)(b * NHEADS + h) * DKV * SEQ;
  const float* biash = bias_d + h * 2047;

  AFrag qa[2];   // dk = 64 -> two K=32 A-fragments
  {
    const unsigned short* qr = Qb + (size_t)(q0 + ln) * DKV;
#pragma unroll
    for (int ks = 0; ks < 2; ++ks) {
      qa[ks].q[0] = *(const uint4*)(qr + ks * 32 + 16 * hh);
      qa[ks].q[1] = *(const uint4*)(qr + ks * 32 + 16 * hh + 8);
    }
  }

  AFrag kf[2][2][2];   // [buf][ntile][kstep]
  AFrag vf[2][4];      // [buf][dvtile]

  auto loadK = [&](int buf, int k0) {
#pragma unroll
    for (int nt = 0; nt < 2; ++nt) {
      const unsigned short* kr = Kb + (size_t)(k0 + nt * 16 + ln) * DKV;
#pragma unroll
      for (int ks = 0; ks < 2; ++ks) {
        kf[buf][nt][ks].q[0] = *(const uint4*)(kr + ks * 32 + 16 * hh);
        kf[buf][nt][ks].q[1] = *(const uint4*)(kr + ks * 32 + 16 * hh + 8);
      }
    }
  };
  auto loadV = [&](int buf, int k0) {
#pragma unroll
    for (int t = 0; t < 4; ++t) {
      const unsigned short* vr =
          Vb + (size_t)(t * 16 + ln) * SEQ + k0 + 16 * hh;
      vf[buf][t].q[0] = *(const uint4*)(vr);
      vf[buf][t].q[1] = *(const uint4*)(vr + 8);
    }
  };

  const v8f vzero = {0.f, 0.f, 0.f, 0.f, 0.f, 0.f, 0.f, 0.f};
  float mrow[8], lrow[8];
  v8f acc[4];
#pragma unroll
  for (int r = 0; r < 8; ++r) { mrow[r] = -3.0e38f; lrow[r] = 0.f; }
#pragma unroll
  for (int t = 0; t < 4; ++t) acc[t] = vzero;

  loadK(0, 0);
  loadV(0, 0);

  for (int k0 = 0; k0 < SEQ; k0 += 32) {
    const int cur = (k0 >> 5) & 1;
    const int nxt = cur ^ 1;

    // ---- scores: two 16x16 tiles covering keys [k0, k0+32) ----
    v8f sc[2] = {vzero, vzero};
#pragma unroll
    for (int nt = 0; nt < 2; ++nt)
#pragma unroll
      for (int ks = 0; ks < 2; ++ks)
        sc[nt] = wmma_bf16(qa[ks], kf[cur][nt][ks], sc[nt]);

    // prefetch next tile's K/V: in flight during softmax + LDS round-trip
    if (k0 + 32 < SEQ) {
      loadK(nxt, k0 + 32);
      loadV(nxt, k0 + 32);
    }

    // ---- bias + mask, per-row tile max ----
    float tmax[8];
#pragma unroll
    for (int r = 0; r < 8; ++r) {
      const int qr = q0 + r + 8 * hh;
      const int kc0 = k0 + ln;
      const int kc1 = k0 + 16 + ln;
      float s0 = sc[0][r] + biash[kc0 - qr + 1023] + mask[qr * SEQ + kc0];
      float s1 = sc[1][r] + biash[kc1 - qr + 1023] + mask[qr * SEQ + kc1];
      sc[0][r] = s0; sc[1][r] = s1;
      tmax[r] = fmaxf(s0, s1);
    }
#pragma unroll
    for (int r = 0; r < 8; ++r) {
      float v = tmax[r];
#pragma unroll
      for (int mm = 1; mm < 16; mm <<= 1) v = fmaxf(v, __shfl_xor(v, mm, 16));
      tmax[r] = v;
    }

    // ---- online softmax update ----
    float scale[8];
#pragma unroll
    for (int r = 0; r < 8; ++r) {
      float mn = fmaxf(mrow[r], tmax[r]);
      scale[r] = __expf(mrow[r] - mn);
      mrow[r] = mn;
    }
#pragma unroll
    for (int r = 0; r < 8; ++r) {
      float p0 = __expf(sc[0][r] - mrow[r]);
      float p1 = __expf(sc[1][r] - mrow[r]);
      sc[0][r] = p0; sc[1][r] = p1;
      float s = p0 + p1;
#pragma unroll
      for (int mm = 1; mm < 16; mm <<= 1) s += __shfl_xor(s, mm, 16);
      lrow[r] = lrow[r] * scale[r] + s;
    }
#pragma unroll
    for (int t = 0; t < 4; ++t)
#pragma unroll
      for (int r = 0; r < 8; ++r) acc[t][r] *= scale[r];

    // ---- P: C-layout -> A-layout via LDS (16 x 32 bf16 tile) ----
    __syncthreads();
#pragma unroll
    for (int r = 0; r < 8; ++r) {
      Plds[(r + 8 * hh) * 32 + ln]      = f2bf(sc[0][r]);
      Plds[(r + 8 * hh) * 32 + 16 + ln] = f2bf(sc[1][r]);
    }
    __syncthreads();
    AFrag pf;
    {
      const unsigned short* pr = Plds + ln * 32;
      pf.q[0] = *(const uint4*)(pr + 16 * hh);
      pf.q[1] = *(const uint4*)(pr + 16 * hh + 8);
    }

    // ---- acc += P @ V ----
#pragma unroll
    for (int t = 0; t < 4; ++t)
      acc[t] = wmma_bf16(pf, vf[cur][t], acc[t]);
  }

  // ---- normalize and store bf16 into AttnOut [B, S, H*dkv] ----
  float inv[8];
#pragma unroll
  for (int r = 0; r < 8; ++r) inv[r] = 1.0f / lrow[r];
#pragma unroll
  for (int t = 0; t < 4; ++t) {
#pragma unroll
    for (int r = 0; r < 8; ++r) {
      const int row = b * SEQ + q0 + r + 8 * hh;
      const int col = h * DKV + t * 16 + ln;
      AttnOut[(size_t)row * DMODEL + col] = f2bf(acc[t][r] * inv[r]);
    }
  }
}

// ---------------------------- output GEMM ----------------------------------
// d_out[4096,1024] fp32 = AttnOut(bf16) @ Wo (Wot transposed bf16).
// Same LDS-staged double-buffered scheme as the QKV GEMM.

__global__ __launch_bounds__(128) void gemm_out_kernel(
    const unsigned short* __restrict__ Abf,
    const unsigned short* __restrict__ Wot,
    float* __restrict__ out) {
  __shared__ __align__(16) unsigned short Blds[2][64 * BROW];

  const int tid = threadIdx.x;
  const int wave = tid >> 5;
  const int lane = tid & 31;
  const int hh = lane >> 4, ln = lane & 15;
  const int m0 = (blockIdx.x * 4 + wave) * 32;
  const int n0 = blockIdx.y * 64;

  const int srow = tid >> 1;
  const int sj = (tid & 1) * 16;
  const unsigned short* wsrc = Wot + (size_t)(n0 + srow) * DMODEL + sj;
  unsigned short* wdst0 = &Blds[0][srow * BROW + sj];
  unsigned short* wdst1 = &Blds[1][srow * BROW + sj];

  const v8f vzero = {0.f, 0.f, 0.f, 0.f, 0.f, 0.f, 0.f, 0.f};
  v8f acc[2][4];
#pragma unroll
  for (int mi = 0; mi < 2; ++mi)
#pragma unroll
    for (int t = 0; t < 4; ++t) acc[mi][t] = vzero;

  const unsigned short* xr[2] = {
      Abf + (size_t)(m0 + ln) * DMODEL,
      Abf + (size_t)(m0 + 16 + ln) * DMODEL};

  AFrag a[2][2];
  uint4 st0, st1;

  auto loadA = [&](int buf, int k0) {
#pragma unroll
    for (int mi = 0; mi < 2; ++mi) {
      a[buf][mi].q[0] = *(const uint4*)(xr[mi] + k0 + 16 * hh);
      a[buf][mi].q[1] = *(const uint4*)(xr[mi] + k0 + 16 * hh + 8);
    }
  };
  auto gloadW = [&](int k0) {
    st0 = *(const uint4*)(wsrc + k0);
    st1 = *(const uint4*)(wsrc + k0 + 8);
  };
  auto storeW = [&](unsigned short* dst) {
    *(uint4*)(dst) = st0;
    *(uint4*)(dst + 8) = st1;
  };

  gloadW(0);
  loadA(0, 0);
  storeW(wdst0);
  __syncthreads();

  for (int k0 = 0; k0 < DMODEL; k0 += 32) {
    const int cur = (k0 >> 5) & 1;
    const int nxt = cur ^ 1;
    const bool more = (k0 + 32) < DMODEL;
    if (more) {
      gloadW(k0 + 32);
      loadA(nxt, k0 + 32);
    }
#pragma unroll
    for (int t = 0; t < 4; ++t) {
      AFrag b;
      const unsigned short* br = &Blds[cur][(t * 16 + ln) * BROW + 16 * hh];
      b.q[0] = *(const uint4*)(br);
      b.q[1] = *(const uint4*)(br + 8);
#pragma unroll
      for (int mi = 0; mi < 2; ++mi)
        acc[mi][t] = wmma_bf16(a[cur][mi], b, acc[mi][t]);
    }
    __syncthreads();
    if (more) {
      storeW(nxt ? wdst1 : wdst0);
      __syncthreads();
    }
  }

#pragma unroll
  for (int mi = 0; mi < 2; ++mi) {
#pragma unroll
    for (int t = 0; t < 4; ++t) {
#pragma unroll
      for (int r = 0; r < 8; ++r) {
        const int row = m0 + 16 * mi + r + 8 * hh;
        const int col = n0 + t * 16 + ln;
        out[(size_t)row * DMODEL + col] = acc[mi][t][r];
      }
    }
  }
}

// ------------------------------- launcher ----------------------------------

extern "C" void kernel_launch(void* const* d_in, const int* in_sizes, int n_in,
                              void* d_out, int out_size, void* d_ws, size_t ws_size,
                              hipStream_t stream) {
  const float* X    = (const float*)d_in[0];  // [4,1024,1024]
  const float* mask = (const float*)d_in[1];  // [1,1,1024,1024]
  const float* wq   = (const float*)d_in[2];
  const float* wk   = (const float*)d_in[3];
  const float* wv   = (const float*)d_in[4];
  const float* wo   = (const float*)d_in[5];
  const float* relb = (const float*)d_in[6];  // [32,16]
  float* out = (float*)d_out;

  char* ws = (char*)d_ws;
  unsigned short* Xbf  = (unsigned short*)(ws + 0);          //  8 MiB
  unsigned short* Wqt  = (unsigned short*)(ws + 8388608);    //  2 MiB
  unsigned short* Wkt  = (unsigned short*)(ws + 10485760);
  unsigned short* Wvt  = (unsigned short*)(ws + 12582912);
  unsigned short* Wot  = (unsigned short*)(ws + 14680064);
  unsigned short* Qd   = (unsigned short*)(ws + 16777216);   //  8 MiB
  unsigned short* Kd   = (unsigned short*)(ws + 25165824);
  unsigned short* Vtd  = (unsigned short*)(ws + 33554432);
  unsigned short* AOd  = (unsigned short*)(ws + 41943040);
  float*          Bias = (float*)        (ws + 50331648);    // 128 KiB

  const int nX = NBATCH * SEQ * DMODEL;  // 4194304
  cvt_bf16_kernel<<<(nX + 255) / 256, 256, 0, stream>>>(X, Xbf, nX);
  {
    dim3 tg(32, 32), tb(32, 8);
    transpose_bf16_kernel<<<tg, tb, 0, stream>>>(wq, Wqt);
    transpose_bf16_kernel<<<tg, tb, 0, stream>>>(wk, Wkt);
    transpose_bf16_kernel<<<tg, tb, 0, stream>>>(wv, Wvt);
    transpose_bf16_kernel<<<tg, tb, 0, stream>>>(wo, Wot);
  }
  bias_kernel<<<(NHEADS * 2047 + 255) / 256, 256, 0, stream>>>(relb, Bias);

  gemm_qkv_kernel<<<dim3(32, 16, 3), 128, 0, stream>>>(Xbf, Wqt, Wkt, Wvt,
                                                       Qd, Kd, Vtd);
  attn_kernel<<<NBATCH * NHEADS * (SEQ / 16), 32, 0, stream>>>(Qd, Kd, Vtd,
                                                               Bias, mask, AOd);
  gemm_out_kernel<<<dim3(32, 16), 128, 0, stream>>>(AOd, Wot, out);
}
